// MaskedAttention_80401787781649
// MI455X (gfx1250) — matrix operand order repaired and checked by reading, not compiled, not run on hardware
//
#include <hip/hip_runtime.h>

// Problem constants (match reference)
#define B_ 2
#define N_ 2048
#define C_ 1024
#define H_ 16
#define D_ 64

typedef __attribute__((ext_vector_type(16))) _Float16 v16h;
typedef __attribute__((ext_vector_type(8)))  _Float16 v8h;
typedef __attribute__((ext_vector_type(8)))  float    v8f;

__device__ __forceinline__ v16h cat16(v8h lo, v8h hi) {
  return __builtin_shufflevector(lo, hi, 0,1,2,3,4,5,6,7,8,9,10,11,12,13,14,15);
}

__device__ __forceinline__ v8f wmma_f16(v16h a, v16h b, v8f c) {
  // D = A(16x32 f16) * B(32x16 f16) + C(16x16 f32)
  return __builtin_amdgcn_wmma_f32_16x16x32_f16(
      /*neg_a=*/false, a, /*neg_b=*/false, b,
      /*c_mod=*/(short)0, c, /*reuse_a=*/false, /*reuse_b=*/false);
}

// ---------------------------------------------------------------------------
// Kernel 1/4: generic GEMM  C[M,Nc] = A[M,K] * W[Nc,K]^T + bias[Nc]
// fp32 in/out, f16 WMMA inner product. M mult of 128, Nc mult of 64, K mult of 32.
// Block = 256 threads (8 waves). Block tile 128x64; wave tile 32x32 (4 WMMA/step,
// 2x fragment reuse in both operands -> 4 wmma : 8 ds_load_b128 per wave-step).
// ---------------------------------------------------------------------------
__global__ __launch_bounds__(256)
void gemm_bias_wmma(const float* __restrict__ A, const float* __restrict__ W,
                    const float* __restrict__ bias, float* __restrict__ Cout,
                    int M, int Ncols, int K)
{
  __shared__ _Float16 As[128][40];  // pitch 40 halves: 16B-aligned rows, conflict-free
  __shared__ _Float16 Bs[64][40];

  const int lane = threadIdx.x;
  const int wave = threadIdx.y;
  const int tid  = wave * 32 + lane;
  const int l15  = lane & 15;
  const int hi   = lane >> 4;
  const int m0   = blockIdx.y * 128;
  const int n0   = blockIdx.x * 64;
  const int m_off = (wave >> 1) * 32;   // 4 wave-rows of 32
  const int n_off = (wave & 1) * 32;    // 2 wave-cols of 32

  v8f acc00 = {}, acc01 = {}, acc10 = {}, acc11 = {};

  for (int k0 = 0; k0 < K; k0 += 32) {
    // Stage A tile (128x32) and W tile (64x32) fp32 -> f16 LDS.
#pragma unroll
    for (int i = 0; i < 8; ++i) {       // A: 2048 float2 / 256 threads
      int idx = tid + i * 256;
      int row = idx >> 4;               // 0..127
      int c2  = idx & 15;
      float2 va = *(const float2*)&A[(size_t)(m0 + row) * K + k0 + c2 * 2];
      As[row][c2 * 2]     = (_Float16)va.x;
      As[row][c2 * 2 + 1] = (_Float16)va.y;
    }
#pragma unroll
    for (int i = 0; i < 4; ++i) {       // W: 1024 float2 / 256 threads
      int idx = tid + i * 256;
      int row = idx >> 4;               // 0..63
      int c2  = idx & 15;
      float2 vb = *(const float2*)&W[(size_t)(n0 + row) * K + k0 + c2 * 2];
      Bs[row][c2 * 2]     = (_Float16)vb.x;
      Bs[row][c2 * 2 + 1] = (_Float16)vb.y;
    }
    __syncthreads();

    // A fragments (16x32): lane<16 row m, K0..15 ; lane>=16: K16..31
    v16h a0 = cat16(*(const v8h*)&As[m_off + l15][hi * 16],
                    *(const v8h*)&As[m_off + l15][hi * 16 + 8]);
    v16h a1 = cat16(*(const v8h*)&As[m_off + 16 + l15][hi * 16],
                    *(const v8h*)&As[m_off + 16 + l15][hi * 16 + 8]);
    // B fragments (32x16): lane n = column n: K0..15 (lane<16) / K16..31 (lane>=16)
    v16h b0 = cat16(*(const v8h*)&Bs[n_off + l15][hi * 16],
                    *(const v8h*)&Bs[n_off + l15][hi * 16 + 8]);
    v16h b1 = cat16(*(const v8h*)&Bs[n_off + 16 + l15][hi * 16],
                    *(const v8h*)&Bs[n_off + 16 + l15][hi * 16 + 8]);
    acc00 = wmma_f16(a0, b0, acc00);
    acc01 = wmma_f16(a0, b1, acc01);
    acc10 = wmma_f16(a1, b0, acc10);
    acc11 = wmma_f16(a1, b1, acc11);
    __syncthreads();
  }

  // C layout: VGPR r, lane<16 -> row r; lane>=16 -> row 8+r; col = lane&15
#pragma unroll
  for (int r = 0; r < 8; ++r) {
    int mg0 = m0 + m_off + hi * 8 + r;
    int ng  = n0 + n_off + l15;
    float bs0 = bias[ng], bs1 = bias[ng + 16];
    Cout[(size_t)mg0 * Ncols + ng]      = acc00[r] + bs0;
    Cout[(size_t)mg0 * Ncols + ng + 16] = acc01[r] + bs1;
    Cout[(size_t)(mg0 + 16) * Ncols + ng]      = acc10[r] + bs0;
    Cout[(size_t)(mg0 + 16) * Ncols + ng + 16] = acc11[r] + bs1;
  }
}

// ---------------------------------------------------------------------------
// Kernel 2: per-head LayerNorm of Q,K + reshape to [B,H,N,D] f16; V -> [B,H,D,N] f16
// One wave handles one (b,n,h) triple; lane owns d = 2*lane, 2*lane+1.
// ---------------------------------------------------------------------------
__device__ __forceinline__ float wsum32(float v) {
#pragma unroll
  for (int m = 1; m < 32; m <<= 1) v += __shfl_xor(v, m, 32);
  return v;
}

__global__ __launch_bounds__(256)
void ln_split(const float* __restrict__ qkv,
              const float* __restrict__ g_q, const float* __restrict__ bt_q,
              const float* __restrict__ g_k, const float* __restrict__ bt_k,
              _Float16* __restrict__ Qh, _Float16* __restrict__ Kh,
              _Float16* __restrict__ Vt)
{
  const int lane = threadIdx.x;
  const int w  = blockIdx.x * 8 + threadIdx.y;  // ((b*N+n)*H + h)
  const int h  = w & (H_ - 1);
  const int bn = w >> 4;                        // b*N + n
  const int b  = bn >> 11;
  const int n  = bn & (N_ - 1);
  const int d0 = lane * 2;

  // qkv layout [B,N,3,H,D]: offset = bn*3*C + s*C + h*D + d
  const size_t base = (size_t)bn * 3 * C_ + (size_t)h * D_;
  float2 q2 = *(const float2*)&qkv[base + d0];
  float2 k2 = *(const float2*)&qkv[base + C_ + d0];
  float2 v2 = *(const float2*)&qkv[base + 2 * C_ + d0];

  // LayerNorm over D=64
  float qmu = wsum32(q2.x + q2.y) * (1.0f / 64.0f);
  float qa = q2.x - qmu, qb = q2.y - qmu;
  float qvar = wsum32(qa * qa + qb * qb) * (1.0f / 64.0f);
  float qr = rsqrtf(qvar + 1e-5f);
  float qo0 = qa * qr * g_q[d0]     + bt_q[d0];
  float qo1 = qb * qr * g_q[d0 + 1] + bt_q[d0 + 1];

  float kmu = wsum32(k2.x + k2.y) * (1.0f / 64.0f);
  float ka = k2.x - kmu, kb = k2.y - kmu;
  float kvar = wsum32(ka * ka + kb * kb) * (1.0f / 64.0f);
  float kr = rsqrtf(kvar + 1e-5f);
  float ko0 = ka * kr * g_k[d0]     + bt_k[d0];
  float ko1 = kb * kr * g_k[d0 + 1] + bt_k[d0 + 1];

  const size_t rowo = ((size_t)(b * H_ + h) * N_ + n) * D_;
  Qh[rowo + d0]     = (_Float16)qo0;
  Qh[rowo + d0 + 1] = (_Float16)qo1;
  Kh[rowo + d0]     = (_Float16)ko0;
  Kh[rowo + d0 + 1] = (_Float16)ko1;

  const size_t vto = ((size_t)(b * H_ + h) * D_ + d0) * N_ + n;
  Vt[vto]      = (_Float16)v2.x;
  Vt[vto + N_] = (_Float16)v2.y;
}

// ---------------------------------------------------------------------------
// Kernel 3: causal flash attention. One wave = 16 queries of one (b,h).
// Q,K f16 [B,H,N,D]; V transposed f16 [B,H,D,N]; output fp32 [B,N,C].
// QK^T: K rows are directly WMMA B-fragments (k-dim = d).
// PV:   Vt rows are directly WMMA B-fragments (k-dim = key).
// ---------------------------------------------------------------------------
__global__ __launch_bounds__(256)
void flash_attn(const _Float16* __restrict__ Qh, const _Float16* __restrict__ Kh,
                const _Float16* __restrict__ Vt, float* __restrict__ Oa)
{
  __shared__ _Float16 Pb[8][16][40];  // per-wave P staging (C-layout -> A-layout)

  const int lane = threadIdx.x;
  const int wave = threadIdx.y;
  const int l15  = lane & 15;
  const int hi   = lane >> 4;
  const int w  = blockIdx.x * 8 + wave;   // 0..4095
  const int bh = w >> 7;                  // b*H + h
  const int q0 = (w & 127) * 16;

  // Q A-fragments for the whole strip (d 0..31 and 32..63)
  const _Float16* Qrow = Qh + ((size_t)bh * N_ + q0 + l15) * D_;
  v16h aq0 = cat16(*(const v8h*)(Qrow + hi * 16),      *(const v8h*)(Qrow + hi * 16 + 8));
  v16h aq1 = cat16(*(const v8h*)(Qrow + 32 + hi * 16), *(const v8h*)(Qrow + 32 + hi * 16 + 8));

  v8f zero = {};
  v8f o[4];
#pragma unroll
  for (int j = 0; j < 4; ++j) o[j] = zero;
  float mi[8], li[8];
#pragma unroll
  for (int r = 0; r < 8; ++r) { mi[r] = -3.0e38f; li[r] = 0.0f; }

  const int kend = q0 + 16;   // causal: keys < q0+16
  for (int kb = 0; kb < kend; kb += 32) {
    v8f s0 = zero, s1 = zero;
    {
      const _Float16* Krow = Kh + ((size_t)bh * N_ + kb + l15) * D_;
      v16h bk0 = cat16(*(const v8h*)(Krow + hi * 16),      *(const v8h*)(Krow + hi * 16 + 8));
      v16h bk1 = cat16(*(const v8h*)(Krow + 32 + hi * 16), *(const v8h*)(Krow + 32 + hi * 16 + 8));
      s0 = wmma_f16(aq0, bk0, s0);
      s0 = wmma_f16(aq1, bk1, s0);
      Krow += 16 * D_;
      bk0 = cat16(*(const v8h*)(Krow + hi * 16),      *(const v8h*)(Krow + hi * 16 + 8));
      bk1 = cat16(*(const v8h*)(Krow + 32 + hi * 16), *(const v8h*)(Krow + 32 + hi * 16 + 8));
      s1 = wmma_f16(aq0, bk0, s1);
      s1 = wmma_f16(aq1, bk1, s1);
    }

    // Online softmax over this 32-key block, per C-layout row (8 rows per half-lane)
    const float sc = 0.125f;  // 1/sqrt(64)
#pragma unroll
    for (int r = 0; r < 8; ++r) {
      const int mrow = q0 + hi * 8 + r;
      float v0 = s0[r] * sc; if (kb + l15 > mrow)      v0 = -3.0e38f;
      float v1 = s1[r] * sc; if (kb + 16 + l15 > mrow) v1 = -3.0e38f;
      float t = fmaxf(v0, v1);
#pragma unroll
      for (int mk = 1; mk < 16; mk <<= 1) t = fmaxf(t, __shfl_xor(t, mk, 32));
      float mn = fmaxf(mi[r], t);
      float f  = __expf(mi[r] - mn);
      float p0 = __expf(v0 - mn);
      float p1 = __expf(v1 - mn);
      float ps = p0 + p1;
#pragma unroll
      for (int mk = 1; mk < 16; mk <<= 1) ps += __shfl_xor(ps, mk, 32);
      li[r] = li[r] * f + ps;
      mi[r] = mn;
      o[0][r] *= f; o[1][r] *= f; o[2][r] *= f; o[3][r] *= f;
      Pb[wave][hi * 8 + r][l15]      = (_Float16)p0;
      Pb[wave][hi * 8 + r][16 + l15] = (_Float16)p1;
    }

    // P (16x32) as A-fragment via LDS
    v16h ap = cat16(*(const v8h*)&Pb[wave][l15][hi * 16],
                    *(const v8h*)&Pb[wave][l15][hi * 16 + 8]);
    // PV: 4 d-tiles of 16; Vt rows are contiguous in keys
#pragma unroll
    for (int j = 0; j < 4; ++j) {
      const _Float16* Vrow = Vt + ((size_t)bh * D_ + j * 16 + l15) * N_ + kb;
      v16h bv = cat16(*(const v8h*)(Vrow + hi * 16), *(const v8h*)(Vrow + hi * 16 + 8));
      o[j] = wmma_f16(ap, bv, o[j]);
    }
  }

  const int b = bh >> 4, h = bh & 15;
#pragma unroll
  for (int r = 0; r < 8; ++r) {
    float inv = 1.0f / li[r];
    float* op = Oa + (size_t)(b * N_ + q0 + hi * 8 + r) * C_ + h * D_ + l15;
    op[0]  = o[0][r] * inv;
    op[16] = o[1][r] * inv;
    op[32] = o[2][r] * inv;
    op[48] = o[3][r] * inv;
  }
}

// ---------------------------------------------------------------------------
// Launcher
// ---------------------------------------------------------------------------
extern "C" void kernel_launch(void* const* d_in, const int* in_sizes, int n_in,
                              void* d_out, int out_size, void* d_ws, size_t ws_size,
                              hipStream_t stream)
{
  (void)in_sizes; (void)n_in; (void)out_size; (void)ws_size;
  const float* x      = (const float*)d_in[0];
  const float* w_qkv  = (const float*)d_in[1];
  const float* b_qkv  = (const float*)d_in[2];
  const float* g_q    = (const float*)d_in[3];
  const float* bt_q   = (const float*)d_in[4];
  const float* g_k    = (const float*)d_in[5];
  const float* bt_k   = (const float*)d_in[6];
  const float* w_proj = (const float*)d_in[7];
  const float* b_proj = (const float*)d_in[8];
  // d_in[9] = attn_mask (causal) handled analytically
  float* out = (float*)d_out;

  char* p = (char*)d_ws;
  float*    qkv  = (float*)p;    p += (size_t)B_ * N_ * 3 * C_ * sizeof(float);     // 50.3 MB
  _Float16* Qh   = (_Float16*)p; p += (size_t)B_ * H_ * N_ * D_ * sizeof(_Float16); // 8.4 MB
  _Float16* Kh   = (_Float16*)p; p += (size_t)B_ * H_ * N_ * D_ * sizeof(_Float16); // 8.4 MB
  _Float16* Vt   = (_Float16*)p; p += (size_t)B_ * H_ * N_ * D_ * sizeof(_Float16); // 8.4 MB
  float*    attn = (float*)p;                                                        // 16.8 MB

  dim3 blk(32, 8);
  // 1) fused QKV projection: (4096x1024) x (3072x1024)^T
  gemm_bias_wmma<<<dim3(3 * C_ / 64, B_ * N_ / 128), blk, 0, stream>>>(
      x, w_qkv, b_qkv, qkv, B_ * N_, 3 * C_, C_);
  // 2) qk-LayerNorm + reshape/transposes
  ln_split<<<dim3(B_ * N_ * H_ / 8), blk, 0, stream>>>(
      qkv, g_q, bt_q, g_k, bt_k, Qh, Kh, Vt);
  // 3) causal flash attention
  flash_attn<<<dim3(B_ * H_ * (N_ / 16) / 8), blk, 0, stream>>>(Qh, Kh, Vt, attn);
  // 4) output projection: (4096x1024) x (1024x1024)^T
  gemm_bias_wmma<<<dim3(C_ / 64, B_ * N_ / 128), blk, 0, stream>>>(
      attn, w_proj, b_proj, out, B_ * N_, C_, C_);
}
